// LSTMcell_12962211300009
// MI455X (gfx1250) — matrix operand order repaired and checked by reading, not compiled, not run on hardware
//
#include <hip/hip_runtime.h>

// Problem dims (compile-time constants from the reference)
#define B_DIM 16384
#define I_DIM 1024
#define H_DIM 1024

// Tiling
#define BM 128          // batch rows per block
#define BN 64           // hidden cols per block (per gate; all 4 gates computed)
#define BK 32           // K slice = one wmma_f32_16x16x32_bf16
#define LDA 40          // LDS row pitch in bf16 elems (80B, 16B aligned, bank-skewed)
#define NSTEP_X (I_DIM / BK)            // 32
#define NSTEP_H (H_DIM / BK)            // 32
#define NSTEP   (NSTEP_X + NSTEP_H)     // 64

typedef __attribute__((ext_vector_type(16))) __bf16 v16bf;
typedef __attribute__((ext_vector_type(8)))  __bf16 v8bf;
typedef __attribute__((ext_vector_type(4)))  __bf16 v4bf;
typedef __attribute__((ext_vector_type(8)))  float  v8f;

union Frag16 { v16bf v; v8bf h[2]; };

// f32 -> bf16 round-to-nearest-even via integer trick (no fptrunc dependence)
__device__ __forceinline__ __bf16 f2bf_rne(float f) {
    unsigned u = __builtin_bit_cast(unsigned, f);
    u += 0x7FFFu + ((u >> 16) & 1u);
    unsigned short s = (unsigned short)(u >> 16);
    return __builtin_bit_cast(__bf16, s);
}

__device__ __forceinline__ float sigmoid_f(float x) {
    return 1.0f / (1.0f + __expf(-x));
}
__device__ __forceinline__ float tanh_f(float x) {
    // tanh(x) = 1 - 2/(exp(2x)+1), saturating and cheap (one v_exp)
    return 1.0f - 2.0f / (__expf(2.0f * x) + 1.0f);
}

// ---------------------------------------------------------------------------
// Pass 1: one-time f32 -> bf16 conversion (memory bound; VALU here is free)
// ---------------------------------------------------------------------------
__global__ __launch_bounds__(256)
void cvt_f32_to_bf16(const float* __restrict__ src, __bf16* __restrict__ dst, int n4) {
    int i = blockIdx.x * blockDim.x + threadIdx.x;
    const int stride = gridDim.x * blockDim.x;
    for (; i < n4; i += stride) {
        const float4 v = ((const float4*)src)[i];
        v4bf b;
        b[0] = f2bf_rne(v.x); b[1] = f2bf_rne(v.y);
        b[2] = f2bf_rne(v.z); b[3] = f2bf_rne(v.w);
        ((v4bf*)dst)[i] = b;
    }
}

// ---------------------------------------------------------------------------
// Pass 2: fused LSTM-cell GEMM. T = __bf16 (preconverted, preferred) or
// T = float (inline-convert fallback when d_ws is too small).
// ---------------------------------------------------------------------------
template <typename T>
__global__ __launch_bounds__(256)
void lstm_fused_wmma(const T* __restrict__ x,  const float* __restrict__ c,
                     const T* __restrict__ h,  const T* __restrict__ Wx,
                     const float* __restrict__ bx, const T* __restrict__ Wh,
                     const float* __restrict__ bh,
                     float* __restrict__ out_o, float* __restrict__ out_c,
                     float* __restrict__ out_h)
{
    // Double-buffered bf16 staging: A (activations) + B (weights, 4 gates)
    __shared__ __bf16 sA[2][BM][LDA];        // 20 KB
    __shared__ __bf16 sB[2][4][BN][LDA];     // 40 KB

    const int tid     = threadIdx.x;
    const int colTile = blockIdx.x % (H_DIM / BN);   // 16
    const int rowTile = blockIdx.x / (H_DIM / BN);   // 128
    const int rowBase = rowTile * BM;
    const int colBase = colTile * BN;

    // Cooperative global -> LDS(bf16) tile load for K-step `step`
    auto load_tile = [&](int step, int buf) {
        const T* aSrc; const T* wSrc; int k0;
        if (step < NSTEP_X) { aSrc = x; wSrc = Wx; k0 = step * BK; }
        else                { aSrc = h; wSrc = Wh; k0 = (step - NSTEP_X) * BK; }

        if constexpr (sizeof(T) == 2) {
            // bf16 source: pure b128 copy, zero conversion VALU in the hot loop.
            // A tile: 128x32 bf16 = 512 x v8bf chunks, 2 per thread
            #pragma unroll
            for (int i = 0; i < 2; ++i) {
                const int idx = tid + i * 256;
                const int r = idx >> 2, c8 = idx & 3;
                const v8bf v = *(const v8bf*)(aSrc + (size_t)(rowBase + r) * 1024 + k0 + c8 * 8);
                *(v8bf*)(&sA[buf][r][c8 * 8]) = v;
            }
            // W tiles: 4 x 64 x 32 bf16 = 1024 x v8bf chunks, 4 per thread
            #pragma unroll
            for (int i = 0; i < 4; ++i) {
                const int idx = tid + i * 256;
                const int g = idx >> 8;          // 256 chunks per gate
                const int rem = idx & 255;
                const int n = rem >> 2, c8 = rem & 3;
                const v8bf v = *(const v8bf*)(wSrc + (size_t)(g * H_DIM + colBase + n) * 1024 + k0 + c8 * 8);
                *(v8bf*)(&sB[buf][g][n][c8 * 8]) = v;
            }
        } else {
            // f32 fallback: convert while staging (only used if ws too small)
            #pragma unroll
            for (int i = 0; i < 4; ++i) {
                const int idx = tid + i * 256;
                const int r = idx >> 3, c4 = idx & 7;
                const float4 v = *(const float4*)(aSrc + (size_t)(rowBase + r) * 1024 + k0 + c4 * 4);
                v4bf bv; bv[0] = f2bf_rne(v.x); bv[1] = f2bf_rne(v.y);
                bv[2] = f2bf_rne(v.z); bv[3] = f2bf_rne(v.w);
                *(v4bf*)(&sA[buf][r][c4 * 4]) = bv;
            }
            #pragma unroll
            for (int i = 0; i < 8; ++i) {
                const int idx = tid + i * 256;
                const int g = idx >> 9;
                const int rem = idx & 511;
                const int n = rem >> 3, c4 = rem & 7;
                const float4 v = *(const float4*)(wSrc + (size_t)(g * H_DIM + colBase + n) * 1024 + k0 + c4 * 4);
                v4bf bv; bv[0] = f2bf_rne(v.x); bv[1] = f2bf_rne(v.y);
                bv[2] = f2bf_rne(v.z); bv[3] = f2bf_rne(v.w);
                *(v4bf*)(&sB[buf][g][n][c4 * 4]) = bv;
            }
        }

        // L2 prefetch two stages ahead (global_prefetch_b8); double-buffer
        // covers stage+1, this covers HBM latency for stage+2.
        const int pfs = step + 2;
        if (pfs < NSTEP) {
            const T* paSrc; const T* pwSrc; int pk0;
            if (pfs < NSTEP_X) { paSrc = x; pwSrc = Wx; pk0 = pfs * BK; }
            else               { paSrc = h; pwSrc = Wh; pk0 = (pfs - NSTEP_X) * BK; }
            // 256 threads cover 128 A-rows (2 threads/row) and 256 W-rows
            __builtin_prefetch(paSrc + (size_t)(rowBase + (tid >> 1)) * 1024 + pk0, 0, 0);
            __builtin_prefetch(pwSrc + (size_t)((tid >> 6) * H_DIM + colBase + (tid & 63)) * 1024 + pk0, 0, 0);
        }
    };

    // Wave mapping: 8 waves, 4 (M) x 2 (N); each wave owns 32x32 per gate
    const int lane = tid & 31;
    const int wid  = tid >> 5;
    const int wm   = (wid & 3) * 32;
    const int wn   = (wid >> 2) * 32;
    const int lr   = lane & 15;     // row (A) / col (B,C) within 16
    const int lh   = lane >> 4;     // half-wave select

    v8f acc[4][2][2];
    #pragma unroll
    for (int g = 0; g < 4; ++g)
        #pragma unroll
        for (int mi = 0; mi < 2; ++mi)
            #pragma unroll
            for (int ni = 0; ni < 2; ++ni)
                acc[g][mi][ni] = (v8f){0.f,0.f,0.f,0.f,0.f,0.f,0.f,0.f};

    load_tile(0, 0);

    for (int step = 0; step < NSTEP; ++step) {
        __syncthreads();
        const int cur = step & 1;
        if (step + 1 < NSTEP) load_tile(step + 1, cur ^ 1);

        // A fragments: ISA layout — lane<16: K{0-7,16-23}; lane>=16: K{8-15,24-31}
        Frag16 af[2];
        #pragma unroll
        for (int mi = 0; mi < 2; ++mi) {
            const __bf16* base = &sA[cur][wm + mi * 16 + lr][lh * 8];
            af[mi].h[0] = *(const v8bf*)(base);
            af[mi].h[1] = *(const v8bf*)(base + 16);
        }
        #pragma unroll
        for (int g = 0; g < 4; ++g) {
            // B fragments: lane = column; lanes 0-15 K{0-7,16-23}, lanes 16-31 K{8-15,24-31}
            Frag16 bf[2];
            #pragma unroll
            for (int ni = 0; ni < 2; ++ni) {
                const __bf16* base = &sB[cur][g][wn + ni * 16 + lr][lh * 8];
                bf[ni].h[0] = *(const v8bf*)(base);
                bf[ni].h[1] = *(const v8bf*)(base + 16);
            }
            #pragma unroll
            for (int mi = 0; mi < 2; ++mi)
                #pragma unroll
                for (int ni = 0; ni < 2; ++ni)
                    acc[g][mi][ni] = __builtin_amdgcn_wmma_f32_16x16x32_bf16(
                        false, af[mi].v, false, bf[ni].v,
                        (short)0, acc[g][mi][ni], false, false);
        }
    }

    // Epilogue: bias + activations + cell/hidden update
    float bias[4][2];
    #pragma unroll
    for (int g = 0; g < 4; ++g)
        #pragma unroll
        for (int ni = 0; ni < 2; ++ni) {
            const int n = colBase + wn + ni * 16 + lr;
            bias[g][ni] = bx[g * H_DIM + n] + bh[g * H_DIM + n];
        }

    #pragma unroll
    for (int mi = 0; mi < 2; ++mi) {
        #pragma unroll
        for (int ni = 0; ni < 2; ++ni) {
            const int n = colBase + wn + ni * 16 + lr;
            #pragma unroll
            for (int r = 0; r < 8; ++r) {
                const int m = rowBase + wm + mi * 16 + r + lh * 8;   // C layout: M = vgpr + 8*half
                const size_t off = (size_t)m * H_DIM + n;
                const float pf = acc[0][mi][ni][r] + bias[0][ni];
                const float pi = acc[1][mi][ni][r] + bias[1][ni];
                const float pk = acc[2][mi][ni][r] + bias[2][ni];
                const float po = acc[3][mi][ni][r] + bias[3][ni];
                const float fg = sigmoid_f(pf);
                const float ig = sigmoid_f(pi);
                const float kg = tanh_f(pk);
                const float og = sigmoid_f(po);
                const float cn = fg * c[off] + ig * kg;
                const float hn = og * tanh_f(cn);
                out_o[off] = og;
                out_c[off] = cn;
                out_h[off] = hn;
            }
        }
    }
}

extern "C" void kernel_launch(void* const* d_in, const int* in_sizes, int n_in,
                              void* d_out, int out_size, void* d_ws, size_t ws_size,
                              hipStream_t stream) {
    (void)in_sizes; (void)n_in; (void)out_size;
    const float* x  = (const float*)d_in[0];
    const float* c  = (const float*)d_in[1];
    const float* h  = (const float*)d_in[2];
    const float* Wx = (const float*)d_in[3];
    const float* bx = (const float*)d_in[4];
    const float* Wh = (const float*)d_in[5];
    const float* bh = (const float*)d_in[6];
    float* out   = (float*)d_out;
    float* out_o = out;
    float* out_c = out + (size_t)B_DIM * H_DIM;
    float* out_h = out + 2 * (size_t)B_DIM * H_DIM;

    const int grid = (H_DIM / BN) * (B_DIM / BM);   // 16 * 128 = 2048 blocks

    const size_t nX = (size_t)B_DIM * I_DIM;        // 16.78M
    const size_t nH = (size_t)B_DIM * H_DIM;        // 16.78M
    const size_t nW = (size_t)4 * H_DIM * I_DIM;    // 4.19M
    const size_t needBytes = (nX + nH + 2 * nW) * 2; // ~80 MB of bf16

    if (ws_size >= needBytes) {
        // Two-pass: convert once (memory bound), then pure-bf16 GEMM.
        __bf16* xb  = (__bf16*)d_ws;
        __bf16* hb  = xb + nX;
        __bf16* Wxb = hb + nH;
        __bf16* Whb = Wxb + nW;
        cvt_f32_to_bf16<<<4096, 256, 0, stream>>>(x,  xb,  (int)(nX / 4));
        cvt_f32_to_bf16<<<4096, 256, 0, stream>>>(h,  hb,  (int)(nH / 4));
        cvt_f32_to_bf16<<<1024, 256, 0, stream>>>(Wx, Wxb, (int)(nW / 4));
        cvt_f32_to_bf16<<<1024, 256, 0, stream>>>(Wh, Whb, (int)(nW / 4));
        lstm_fused_wmma<__bf16><<<grid, 256, 0, stream>>>(
            xb, c, hb, Wxb, bx, Whb, bh, out_o, out_c, out_h);
    } else {
        // Fallback: inline conversion inside the GEMM loader.
        lstm_fused_wmma<float><<<grid, 256, 0, stream>>>(
            x, c, h, Wx, bx, Wh, bh, out_o, out_c, out_h);
    }
}
